// BiDAF_40218073760087
// MI455X (gfx1250) — matrix operand order repaired and checked by reading, not compiled, not run on hardware
//
#include <hip/hip_runtime.h>
#include <hip/hip_bf16.h>
#include <stdint.h>

// ---------------- problem constants ----------------
constexpr int Bx  = 32;
constexpr int LSx = 512;
constexpr int LTx = 512;
constexpr int Dx  = 400;          // embedding dim
constexpr int DP  = 416;          // Dx padded to multiple of 32 (13 K-chunks)
constexpr int OUTW = 4 * Dx;      // 1600
constexpr int TI  = 128;          // i-tile rows per workgroup (8 waves x 16 rows)
constexpr int TJ  = 64;           // j-tile rows per iteration
constexpr int NTHREADS = 256;     // 8 wave32

typedef __attribute__((ext_vector_type(16))) _Float16 v16h;
typedef __attribute__((ext_vector_type(8)))  _Float16 v8h;
typedef __attribute__((ext_vector_type(8)))  float    v8f;

union Frag16 { v16h v; v8h h[2]; };

__device__ inline v8f vzero8() {
  v8f z;
#pragma unroll
  for (int k = 0; k < 8; ++k) z[k] = 0.0f;
  return z;
}

// LDS layout (element counts)
constexpr int SA_E    = TI * DP;   // S*wm tile, f16
constexpr int ST_E    = TJ * DP;   // T tile (pass1: [j][d]; pass2: [d][j], Dx*TJ <= ST_E)
constexpr int SATTN_E = TI * LTx;  // sim -> attn buffer, f16
constexpr size_t SMEM1 = (size_t)(SA_E + ST_E + SATTN_E) * sizeof(_Float16)
                       + (size_t)(TJ + TI + TI + NTHREADS) * sizeof(float);

// =====================================================================
// Kernel 1: fused  sim = (S*wm)·T^T (+t_proj)  -> softmax -> attn·T
// writes out[:, :, 0:3D] and the per-row max m[i] (+s_proj) to workspace.
// =====================================================================
__global__ void bidaf_attn_kernel(const float* __restrict__ S,
                                  const float* __restrict__ T,
                                  const float* __restrict__ w,
                                  float* __restrict__ out,
                                  float* __restrict__ mOut)
{
  extern __shared__ char smem[];
  _Float16* sA    = (_Float16*)smem;          // [TI][DP]
  _Float16* sT    = sA + SA_E;                // [TJ][DP]  (pass2: [Dx][TJ])
  _Float16* sAttn = sT + ST_E;                // [TI][LTx]
  float* fb     = (float*)(sAttn + SATTN_E);
  float* tProj  = fb;                         // [TJ]
  float* sProj  = fb + TJ;                    // [TI]
  float* rowSum = fb + TJ + TI;               // [TI]
  float* red    = fb + TJ + 2 * TI;           // [NTHREADS]

  const int b   = blockIdx.y;
  const int i0  = blockIdx.x * TI;
  const int tid = threadIdx.x;
  const int lane = tid & 31;
  const int wv   = tid >> 5;      // wave id: rows 16*wv .. 16*wv+15
  const int hi   = lane >> 4;     // lane half
  const int lo   = lane & 15;

  const float* ws = w;
  const float* wt = w + Dx;
  const float* wm = w + 2 * Dx;
  const float* Sb = S + (size_t)b * LSx * Dx;
  const float* Tb = T + (size_t)b * LTx * Dx;

  // ---- stage S' = S*wm tile as f16 (pad K with zeros); compute s_proj ----
  for (int idx = tid; idx < TI * DP; idx += NTHREADS) {
    int r = idx / DP, d = idx - r * DP;
    float v = (d < Dx) ? Sb[(size_t)(i0 + r) * Dx + d] * wm[d] : 0.0f;
    sA[idx] = (_Float16)v;
  }
  if (tid < TI) {
    const float* row = Sb + (size_t)(i0 + tid) * Dx;
    float acc = 0.0f;
    for (int d = 0; d < Dx; ++d) acc += row[d] * ws[d];
    sProj[tid] = acc;
  }
  __syncthreads();

  // =========== pass 1: sim tiles -> sAttn (f16, t_proj added) ===========
  for (int jt = 0; jt < LTx / TJ; ++jt) {
    // stage T j-tile as f16 (row major, padded K); compute t_proj
    for (int idx = tid; idx < TJ * DP; idx += NTHREADS) {
      int r = idx / DP, d = idx - r * DP;
      float v = (d < Dx) ? Tb[(size_t)(jt * TJ + r) * Dx + d] : 0.0f;
      sT[idx] = (_Float16)v;
    }
    if (tid < TJ) {
      const float* row = Tb + (size_t)(jt * TJ + tid) * Dx;
      float acc = 0.0f;
      for (int d = 0; d < Dx; ++d) acc += row[d] * wt[d];
      tProj[tid] = acc;
    }
    __syncthreads();

    v8f acc[4];
#pragma unroll
    for (int c = 0; c < 4; ++c) acc[c] = vzero8();

#pragma unroll
    for (int k = 0; k < DP / 32; ++k) {
      // A frag: row = 16*wv + lo; 16 contiguous K at 32k+16*hi
      Frag16 a;
      const _Float16* ar = sA + (size_t)(16 * wv + lo) * DP + 32 * k;
      a.h[0] = *(const v8h*)(ar + 16 * hi);
      a.h[1] = *(const v8h*)(ar + 16 * hi + 8);
#pragma unroll
      for (int c = 0; c < 4; ++c) {
        // B frag: col j = 16c+lo, 16 contiguous K at 32k+16*hi
        Frag16 bf;
        const _Float16* br = sT + (size_t)(16 * c + lo) * DP + 32 * k + 16 * hi;
        bf.h[0] = *(const v8h*)(br);
        bf.h[1] = *(const v8h*)(br + 8);
        acc[c] = __builtin_amdgcn_wmma_f32_16x16x32_f16(
            false, a.v, false, bf.v, (short)0, acc[c], false, false);
      }
    }

    // add t_proj (row-constant s_proj skipped: softmax-invariant) and store
#pragma unroll
    for (int c = 0; c < 4; ++c) {
      float tp = tProj[16 * c + lo];
#pragma unroll
      for (int r = 0; r < 8; ++r) {
        int row = 16 * wv + 8 * hi + r;
        int col = jt * TJ + 16 * c + lo;
        sAttn[(size_t)row * LTx + col] = (_Float16)(acc[c][r] + tp);
      }
    }
    __syncthreads();
  }

  // =========== softmax over j (in-place in sAttn); m[i] to global =======
  {
    const int row  = tid >> 1;       // 2 threads per row
    const int half = tid & 1;
    _Float16* arow = sAttn + (size_t)row * LTx + half * 256;
    float mx = -3.0e38f;
    for (int j = 0; j < 256; ++j) mx = fmaxf(mx, (float)arow[j]);
    red[tid] = mx;
    __syncthreads();
    float rmx = fmaxf(red[2 * row], red[2 * row + 1]);
    __syncthreads();
    float sm = 0.0f;
    for (int j = 0; j < 256; ++j) {
      float p = __expf((float)arow[j] - rmx);
      sm += p;
      arow[j] = (_Float16)p;
    }
    red[tid] = sm;
    __syncthreads();
    if (half == 0) {
      rowSum[row] = red[2 * row] + red[2 * row + 1];
      mOut[(size_t)b * LSx + i0 + row] = rmx + sProj[row];
    }
    __syncthreads();
  }

  // =========== pass 2: ST = attn · T  (K = j) ===========================
  v8f st[25];
#pragma unroll
  for (int n = 0; n < 25; ++n) st[n] = vzero8();

  for (int jt = 0; jt < LTx / TJ; ++jt) {
    // stage T j-tile transposed: sT[d][j], 16-contiguous-K per B lane-half
    for (int idx = tid; idx < TJ * Dx; idx += NTHREADS) {
      int j = idx / Dx, d = idx - j * Dx;
      sT[d * TJ + j] = (_Float16)Tb[(size_t)(jt * TJ + j) * Dx + d];
    }
    __syncthreads();
#pragma unroll
    for (int kc = 0; kc < 2; ++kc) {
      Frag16 a;
      const _Float16* ar = sAttn + (size_t)(16 * wv + lo) * LTx + jt * TJ + 32 * kc;
      a.h[0] = *(const v8h*)(ar + 16 * hi);
      a.h[1] = *(const v8h*)(ar + 16 * hi + 8);
#pragma unroll
      for (int n = 0; n < 25; ++n) {
        Frag16 bf;
        const _Float16* br = sT + (size_t)(16 * n + lo) * TJ + 32 * kc + 16 * hi;
        bf.h[0] = *(const v8h*)(br);
        bf.h[1] = *(const v8h*)(br + 8);
        st[n] = __builtin_amdgcn_wmma_f32_16x16x32_f16(
            false, a.v, false, bf.v, (short)0, st[n], false, false);
      }
    }
    __syncthreads();
  }

  // =========== epilogue: normalize + write out[:, 0:3D] =================
  float* outB = out + ((size_t)b * LSx + i0) * OUTW;
#pragma unroll
  for (int n = 0; n < 25; ++n) {
    int d = 16 * n + lo;
#pragma unroll
    for (int r = 0; r < 8; ++r) {
      int row = 16 * wv + 8 * hi + r;
      float stv = st[n][r] / rowSum[row];
      float sv  = Sb[(size_t)(i0 + row) * Dx + d];
      float* orow = outB + (size_t)row * OUTW;
      orow[d]           = sv;         // S
      orow[Dx + d]      = stv;        // source_target
      orow[2 * Dx + d]  = sv * stv;   // S * source_target
    }
  }
}

// =====================================================================
// Kernel 2: per-batch softmax over m[i], TS[b,d] = sum_i w[i] * S[b,i,d]
// =====================================================================
__global__ void bidaf_ts_kernel(const float* __restrict__ S,
                                const float* __restrict__ mIn,
                                float* __restrict__ TS)
{
  __shared__ float red[NTHREADS];
  __shared__ float wgt[LSx];
  const int b = blockIdx.x, tid = threadIdx.x;

  float v0 = mIn[(size_t)b * LSx + tid];
  float v1 = mIn[(size_t)b * LSx + 256 + tid];
  red[tid] = fmaxf(v0, v1);
  __syncthreads();
  for (int s = 128; s > 0; s >>= 1) {
    if (tid < s) red[tid] = fmaxf(red[tid], red[tid + s]);
    __syncthreads();
  }
  float mx = red[0];
  __syncthreads();
  float e0 = __expf(v0 - mx), e1 = __expf(v1 - mx);
  red[tid] = e0 + e1;
  __syncthreads();
  for (int s = 128; s > 0; s >>= 1) {
    if (tid < s) red[tid] += red[tid + s];
    __syncthreads();
  }
  float inv = 1.0f / red[0];
  wgt[tid] = e0 * inv;
  wgt[256 + tid] = e1 * inv;
  __syncthreads();

  const int d0 = tid, d1 = 256 + tid;
  float a0 = 0.0f, a1 = 0.0f;
  for (int i = 0; i < LSx; ++i) {
    const float* row = S + ((size_t)b * LSx + i) * Dx;
    float wv = wgt[i];
    a0 += wv * row[d0];
    if (d1 < Dx) a1 += wv * row[d1];
  }
  TS[(size_t)b * Dx + d0] = a0;
  if (d1 < Dx) TS[(size_t)b * Dx + d1] = a1;
}

// =====================================================================
// Kernel 3: out[b,i,3D+d] = S[b,i,d] * TS[b,d]
// =====================================================================
__global__ void bidaf_bcast_kernel(const float* __restrict__ S,
                                   const float* __restrict__ TS,
                                   float* __restrict__ out)
{
  int idx = blockIdx.x * blockDim.x + threadIdx.x;
  if (idx >= Bx * LSx * Dx) return;
  int d  = idx % Dx;
  int bi = idx / Dx;          // b*LSx + i
  int b  = bi / LSx;
  out[(size_t)bi * OUTW + 3 * Dx + d] = S[idx] * TS[(size_t)b * Dx + d];
}

extern "C" void kernel_launch(void* const* d_in, const int* in_sizes, int n_in,
                              void* d_out, int out_size, void* d_ws, size_t ws_size,
                              hipStream_t stream) {
  const float* S = (const float*)d_in[0];   // (B, LS, D)
  const float* T = (const float*)d_in[1];   // (B, LT, D)
  const float* w = (const float*)d_in[2];   // (3D,)
  float* out = (float*)d_out;               // (B, LS, 4D)

  float* mBuf = (float*)d_ws;               // B*LS row maxes
  float* TS   = mBuf + Bx * LSx;            // B*D target_source

  bidaf_attn_kernel<<<dim3(LSx / TI, Bx), NTHREADS, SMEM1, stream>>>(S, T, w, out, mBuf);
  bidaf_ts_kernel<<<Bx, NTHREADS, 0, stream>>>(S, mBuf, TS);
  int total = Bx * LSx * Dx;
  bidaf_bcast_kernel<<<(total + NTHREADS - 1) / NTHREADS, NTHREADS, 0, stream>>>(S, TS, out);
}
